// GNN_42666205118904
// MI455X (gfx1250) — matrix operand order repaired and checked by reading, not compiled, not run on hardware
//
#include <hip/hip_runtime.h>
#include <hip/hip_bf16.h>

typedef _Float16 v16h __attribute__((ext_vector_type(16)));
typedef _Float16 v8h  __attribute__((ext_vector_type(8)));
typedef float    v8f  __attribute__((ext_vector_type(8)));

#define WPB 8  // waves per block (256 threads, wave32)

__device__ __forceinline__ v8f wmma16(v16h a, v16h b, v8f c) {
  // D = A(16x32 f16) * B(32x16 f16) + C(16x16 f32)
  return __builtin_amdgcn_wmma_f32_16x16x32_f16(false, a, false, b, (short)0, c, false, false);
}

__device__ __forceinline__ v8f bcast8(float v) {
  v8f r;
#pragma unroll
  for (int i = 0; i < 8; ++i) r[i] = v;
  return r;
}

__device__ __forceinline__ v16h join16(v8h lo, v8h hi) {
  v16h r;
#pragma unroll
  for (int i = 0; i < 8; ++i) { r[i] = lo[i]; r[i + 8] = hi[i]; }
  return r;
}

// A fragment (16x32 f16). rowptr = this lane's row (row = lane&15), row-major stride 64.
// lanes 0-15 hold K = kb+0..7 and kb+16..23 ; lanes 16-31 hold K = kb+8..15 and kb+24..31.
__device__ __forceinline__ v16h a_frag(const _Float16* rowptr, int kb, int kg) {
  const _Float16* p = rowptr + kb + kg * 8;
  return join16(*(const v8h*)p, *(const v8h*)(p + 16));
}

// B fragment (32x16 f16) from TRANSPOSED weights WT[n][k]: lane col = lane&15,
// lanes 0-15 hold K = kb+0..7 and kb+16..23 ; lanes 16-31 hold K = kb+8..15 and kb+24..31.
__device__ __forceinline__ v16h b_frag(const _Float16* colptr, int kb, int kg) {
  const _Float16* p = colptr + kb + kg * 8;
  return join16(*(const v8h*)p, *(const v8h*)(p + 16));
}

__device__ __forceinline__ float gelu_f(float v) {
  return 0.5f * v * (1.0f + erff(v * 0.7071067811865475f));
}

__device__ __forceinline__ float half16_reduce(float v) {
  // sum across the 16 lanes of each wave half (matches WMMA D layout columns)
  v += __shfl_xor(v, 1, 32);
  v += __shfl_xor(v, 2, 32);
  v += __shfl_xor(v, 4, 32);
  v += __shfl_xor(v, 8, 32);
  return v;
}

__device__ __forceinline__ void atomic_max_f(float* addr, float v) {
  if (v >= 0.0f) atomicMax((int*)addr, __float_as_int(v));
  else           atomicMin((unsigned int*)addr, (unsigned int)__float_as_int(v));
}

__device__ __forceinline__ void wave_lds_fence() {
  asm volatile("s_wait_dscnt 0x0" ::: "memory");
}

// ---------------- weight transpose + f16 convert ----------------
__global__ void transpose_w_kernel(const float* __restrict__ W, _Float16* __restrict__ out,
                                   int K, int N, int strideK) {
  int i = blockIdx.x * blockDim.x + threadIdx.x;
  int total = N * strideK;
  if (i >= total) return;
  int n = i / strideK, k = i - n * strideK;
  out[i] = (k < K) ? (_Float16)W[(size_t)k * N + n] : (_Float16)0.0f;
}

// ---------------- N0: h = LN(relu(x @ adapt_w + b)) ----------------
__global__ __launch_bounds__(256) void n0_kernel(
    const float* __restrict__ x, const _Float16* __restrict__ aT,
    const float* __restrict__ ab, const float* __restrict__ g0,
    const float* __restrict__ b0, float* __restrict__ h, int nN, int ntiles) {
  __shared__ __align__(16) _Float16 s_aT[64 * 128];
  for (int i = threadIdx.x; i < 64 * 128; i += 256) s_aT[i] = aT[i];
  __syncthreads();
  const int lane = threadIdx.x & 31, wid = threadIdx.x >> 5;
  const int colLo = lane & 15, kg = lane >> 4;
  float abc[4], gc[4], bc[4];
#pragma unroll
  for (int nt = 0; nt < 4; ++nt) {
    int n = colLo + 16 * nt;
    abc[nt] = ab[n]; gc[nt] = g0[n]; bc[nt] = b0[n];
  }
  const int slot = blockIdx.x * WPB + wid, nslots = gridDim.x * WPB;
  for (int tile = slot; tile < ntiles; tile += nslots) {
    const int base = tile * 16;
    int arow = base + colLo; if (arow >= nN) arow = nN - 1;
    const float* xr = x + (size_t)arow * 128;
    // prefetch next tile's x row for this lane
    {
      int tnext = tile + nslots;
      if (tnext < ntiles) {
        int nr = tnext * 16 + colLo; if (nr >= nN) nr = nN - 1;
        __builtin_prefetch(x + (size_t)nr * 128, 0, 1);
      }
    }
    v16h a[4];
#pragma unroll
    for (int ks = 0; ks < 4; ++ks) {
      const float* p = xr + ks * 32 + kg * 8;
      v16h f;
#pragma unroll
      for (int i = 0; i < 8; ++i) { f[i] = (_Float16)p[i]; f[i + 8] = (_Float16)p[i + 16]; }
      a[ks] = f;
    }
    v8f acc[4];
#pragma unroll
    for (int nt = 0; nt < 4; ++nt) {
      v8f c = bcast8(abc[nt]);
      const _Float16* col = &s_aT[(colLo + 16 * nt) * 128];
#pragma unroll
      for (int ks = 0; ks < 4; ++ks) c = wmma16(a[ks], b_frag(col, ks * 32, kg), c);
      acc[nt] = c;
    }
#pragma unroll
    for (int nt = 0; nt < 4; ++nt)
#pragma unroll
      for (int r = 0; r < 8; ++r) acc[nt][r] = fmaxf(acc[nt][r], 0.0f);
    float mean[8], inv[8];
#pragma unroll
    for (int r = 0; r < 8; ++r)
      mean[r] = half16_reduce(acc[0][r] + acc[1][r] + acc[2][r] + acc[3][r]) * (1.0f / 64.0f);
#pragma unroll
    for (int r = 0; r < 8; ++r) {
      float d0 = acc[0][r] - mean[r], d1 = acc[1][r] - mean[r];
      float d2 = acc[2][r] - mean[r], d3 = acc[3][r] - mean[r];
      float var = half16_reduce(d0 * d0 + d1 * d1 + d2 * d2 + d3 * d3) * (1.0f / 64.0f);
      inv[r] = rsqrtf(var + 1e-5f);
    }
#pragma unroll
    for (int nt = 0; nt < 4; ++nt) {
      int n = colLo + 16 * nt;
#pragma unroll
      for (int r = 0; r < 8; ++r) {
        int node = base + r + 8 * kg;
        float o = (acc[nt][r] - mean[r]) * inv[r] * gc[nt] + bc[nt];
        if (node < nN) h[(size_t)node * 64 + n] = o;
      }
    }
  }
}

// ---------------- N1: hn = LN(h); hn16 (f16) out; q = hn @ Wq + bq ----------------
__global__ __launch_bounds__(256) void n1_kernel(
    const float* __restrict__ h, const float* __restrict__ g,
    const float* __restrict__ b, const _Float16* __restrict__ wqT,
    const float* __restrict__ bq, _Float16* __restrict__ hn16,
    float* __restrict__ q, int nN, int ntiles) {
  __shared__ __align__(16) _Float16 s_wq[64 * 64];
  __shared__ __align__(16) _Float16 s_stage[WPB][16 * 64];
  for (int i = threadIdx.x; i < 64 * 64; i += 256) s_wq[i] = wqT[i];
  __syncthreads();
  const int lane = threadIdx.x & 31, wid = threadIdx.x >> 5;
  const int colLo = lane & 15, kg = lane >> 4;
  float gc[4], bc[4], bqc[4];
#pragma unroll
  for (int nt = 0; nt < 4; ++nt) {
    int n = colLo + 16 * nt;
    gc[nt] = g[n]; bc[nt] = b[n]; bqc[nt] = bq[n];
  }
  const int slot = blockIdx.x * WPB + wid, nslots = gridDim.x * WPB;
  for (int tile = slot; tile < ntiles; tile += nslots) {
    const int base = tile * 16;
    float v[4][8];
#pragma unroll
    for (int nt = 0; nt < 4; ++nt)
#pragma unroll
      for (int r = 0; r < 8; ++r) {
        int node = base + r + 8 * kg; if (node >= nN) node = nN - 1;
        v[nt][r] = h[(size_t)node * 64 + colLo + 16 * nt];
      }
    float mean[8], inv[8];
#pragma unroll
    for (int r = 0; r < 8; ++r)
      mean[r] = half16_reduce(v[0][r] + v[1][r] + v[2][r] + v[3][r]) * (1.0f / 64.0f);
#pragma unroll
    for (int r = 0; r < 8; ++r) {
      float d0 = v[0][r] - mean[r], d1 = v[1][r] - mean[r];
      float d2 = v[2][r] - mean[r], d3 = v[3][r] - mean[r];
      float var = half16_reduce(d0 * d0 + d1 * d1 + d2 * d2 + d3 * d3) * (1.0f / 64.0f);
      inv[r] = rsqrtf(var + 1e-5f);
    }
#pragma unroll
    for (int nt = 0; nt < 4; ++nt) {
      int n = colLo + 16 * nt;
#pragma unroll
      for (int r = 0; r < 8; ++r) {
        int node = base + r + 8 * kg;
        float o = (v[nt][r] - mean[r]) * inv[r] * gc[nt] + bc[nt];
        _Float16 oh = (_Float16)o;
        s_stage[wid][(r + 8 * kg) * 64 + n] = oh;
        if (node < nN) hn16[(size_t)node * 64 + n] = oh;
      }
    }
    wave_lds_fence();
    const _Float16* arow = &s_stage[wid][colLo * 64];
    v16h a0 = a_frag(arow, 0, kg), a1 = a_frag(arow, 32, kg);
#pragma unroll
    for (int nt = 0; nt < 4; ++nt) {
      const _Float16* col = &s_wq[(colLo + 16 * nt) * 64];
      v8f c = bcast8(bqc[nt]);
      c = wmma16(a0, b_frag(col, 0, kg), c);
      c = wmma16(a1, b_frag(col, 32, kg), c);
#pragma unroll
      for (int r = 0; r < 8; ++r) {
        int node = base + r + 8 * kg;
        if (node < nN) q[(size_t)node * 64 + colLo + 16 * nt] = c[r];
      }
    }
    wave_lds_fence();
  }
}

// ---------------- init m/den/aggr ----------------
__global__ void init_kernel(float* __restrict__ m, float* __restrict__ den,
                            float* __restrict__ aggr, int nN) {
  int i = blockIdx.x * blockDim.x + threadIdx.x;
  if (i < nN * 64) aggr[i] = 0.0f;
  if (i < nN) { m[i] = -3.402823466e38f; den[i] = 0.0f; }
}

// ---------------- Edge passes ----------------
// PASS 1: xjt -> k -> att ; att[e] store + atomicMax segment max
// PASS 2: xjt -> v ; scatter exp(att-m)*v into aggr, exp into den
template <int PASS>
__global__ __launch_bounds__(256) void edge_kernel(
    const float* __restrict__ et, const float* __restrict__ es,
    const int* __restrict__ srcI, const int* __restrict__ dstI,
    const _Float16* __restrict__ hn16, const float* __restrict__ q,
    const _Float16* __restrict__ wtT, const _Float16* __restrict__ w1T,
    const _Float16* __restrict__ w2T, const float* __restrict__ btp,
    const float* __restrict__ b1p, const float* __restrict__ b2p,
    float* __restrict__ att, float* __restrict__ mbuf,
    float* __restrict__ den, float* __restrict__ aggr, int nE, int ntiles) {
  __shared__ __align__(16) _Float16 s_wt[64 * 72];   // WtT padded (65 -> 72)
  __shared__ __align__(16) _Float16 s_w1[64 * 64];
  __shared__ __align__(16) _Float16 s_w2[64 * 64];
  __shared__ __align__(16) _Float16 s_xjt[WPB][16 * 64];
  __shared__ float s_t[WPB][16];
  __shared__ float s_es[WPB][16];
  __shared__ int   s_src[WPB][16];
  __shared__ int   s_dst[WPB][16];
  __shared__ float s_ex[WPB][16];

  for (int i = threadIdx.x; i < 64 * 72; i += 256) s_wt[i] = wtT[i];
  for (int i = threadIdx.x; i < 64 * 64; i += 256) { s_w1[i] = w1T[i]; s_w2[i] = w2T[i]; }
  __syncthreads();

  const int lane = threadIdx.x & 31, wid = threadIdx.x >> 5;
  const int colLo = lane & 15, kg = lane >> 4;
  float btc[4], b1c[4], b2c[4], divc[4], w65c[4];
  bool evn[4];
#pragma unroll
  for (int nt = 0; nt < 4; ++nt) {
    int n = colLo + 16 * nt;
    btc[nt] = btp[n]; b1c[nt] = b1p[n]; b2c[nt] = b2p[n];
    divc[nt] = exp2f(-(float)(2 * (n >> 1)) * (13.287712379549449f / 64.0f));
    evn[nt] = ((n & 1) == 0);
    w65c[nt] = (float)s_wt[n * 72 + 64];
  }
  const int slot = blockIdx.x * WPB + wid, nslots = gridDim.x * WPB;
  for (int tile = slot; tile < ntiles; tile += nslots) {
    const int e0 = tile * 16;
    if (lane < 16) {
      int ee = e0 + lane; if (ee >= nE) ee = nE - 1;
      s_t[wid][lane] = et[ee];
      s_es[wid][lane] = es[ee];
      int d = dstI[ee];
      s_src[wid][lane] = srcI[ee];
      s_dst[wid][lane] = d;
      if (PASS == 2) s_ex[wid][lane] = __expf(att[ee] - mbuf[d]);
      // software prefetch: pull next tile's A rows (hn16 gather) toward the WGP
      int tnext = tile + nslots;
      if (tnext < ntiles) {
        int en = tnext * 16 + lane; if (en >= nE) en = nE - 1;
        __builtin_prefetch(hn16 + (size_t)srcI[en] * 64, 0, 1);
      }
    }
    wave_lds_fence();
    // ---- xjt = gelu([hn[src], t] @ Wt + bt) + te ----
    const _Float16* arow = hn16 + (size_t)s_src[wid][colLo] * 64;
    v16h a0 = a_frag(arow, 0, kg), a1 = a_frag(arow, 32, kg);
    v8f acc[4];
#pragma unroll
    for (int nt = 0; nt < 4; ++nt) {
      const _Float16* col = &s_wt[(colLo + 16 * nt) * 72];
      v8f c = bcast8(btc[nt]);
      c = wmma16(a0, b_frag(col, 0, kg), c);
      c = wmma16(a1, b_frag(col, 32, kg), c);
      acc[nt] = c;
    }
#pragma unroll
    for (int nt = 0; nt < 4; ++nt) {
      int n = colLo + 16 * nt;
#pragma unroll
      for (int r = 0; r < 8; ++r) {
        int rr = r + 8 * kg;
        float t = s_t[wid][rr];
        float vv = acc[nt][r] + t * w65c[nt];          // rank-1 t column
        vv = gelu_f(vv);
        float pt = t * 200.0f * divc[nt];
        vv += evn[nt] ? __sinf(pt) : __cosf(pt);       // temporal encoding
        s_xjt[wid][rr * 64 + n] = (_Float16)vv;
      }
    }
    wave_lds_fence();
    // ---- re-feed xjt as A for blended K (pass1) / V (pass2) ----
    const _Float16* xrow = &s_xjt[wid][colLo * 64];
    v16h xa0 = a_frag(xrow, 0, kg), xa1 = a_frag(xrow, 32, kg);
    v8f outv[4];
#pragma unroll
    for (int nt = 0; nt < 4; ++nt) {
      const _Float16* c1p = &s_w1[(colLo + 16 * nt) * 64];
      const _Float16* c2p = &s_w2[(colLo + 16 * nt) * 64];
      v8f c1 = bcast8(b1c[nt]);
      c1 = wmma16(xa0, b_frag(c1p, 0, kg), c1);
      c1 = wmma16(xa1, b_frag(c1p, 32, kg), c1);
      v8f c2 = bcast8(b2c[nt]);
      c2 = wmma16(xa0, b_frag(c2p, 0, kg), c2);
      c2 = wmma16(xa1, b_frag(c2p, 32, kg), c2);
      v8f o;
#pragma unroll
      for (int r = 0; r < 8; ++r) {
        float e = s_es[wid][r + 8 * kg];
        o[r] = e * c1[r] + (1.0f - e) * c2[r];
      }
      outv[nt] = o;
    }
    if (PASS == 1) {
      float p[8];
#pragma unroll
      for (int r = 0; r < 8; ++r) {
        int rr = r + 8 * kg;
        const float* qrow = q + (size_t)s_dst[wid][rr] * 64;
        float s = 0.0f;
#pragma unroll
        for (int nt = 0; nt < 4; ++nt) s += outv[nt][r] * qrow[colLo + 16 * nt];
        p[r] = half16_reduce(s);
      }
#pragma unroll
      for (int r = 0; r < 8; ++r) {
        if ((lane & 15) == r) {
          int rr = r + 8 * kg;
          int ee = e0 + rr;
          if (ee < nE) {
            float a = p[r] * 0.125f;                   // / sqrt(64)
            att[ee] = a;
            atomic_max_f(&mbuf[s_dst[wid][rr]], a);
          }
        }
      }
    } else {
#pragma unroll
      for (int nt = 0; nt < 4; ++nt) {
        int n = colLo + 16 * nt;
#pragma unroll
        for (int r = 0; r < 8; ++r) {
          int rr = r + 8 * kg;
          int ee = e0 + rr;
          if (ee < nE)
            atomicAdd(&aggr[(size_t)s_dst[wid][rr] * 64 + n], s_ex[wid][rr] * outv[nt][r]);
        }
      }
#pragma unroll
      for (int r = 0; r < 8; ++r) {
        if ((lane & 15) == r) {
          int rr = r + 8 * kg;
          int ee = e0 + rr;
          if (ee < nE) atomicAdd(&den[s_dst[wid][rr]], s_ex[wid][rr]);
        }
      }
    }
    wave_lds_fence();
  }
}

// ---------------- N2: h += gelu(aggr / (den + eps)) ----------------
__global__ void n2_kernel(float* __restrict__ h, const float* __restrict__ aggr,
                          const float* __restrict__ den, int nN) {
  int i = blockIdx.x * blockDim.x + threadIdx.x;
  if (i >= nN * 64) return;
  float d = den[i >> 6];
  h[i] += gelu_f(aggr[i] / (d + 1e-16f));
}

extern "C" void kernel_launch(void* const* d_in, const int* in_sizes, int n_in,
                              void* d_out, int out_size, void* d_ws, size_t ws_size,
                              hipStream_t stream) {
  (void)n_in; (void)out_size; (void)ws_size;
  const float* x         = (const float*)d_in[0];
  const float* edge_time = (const float*)d_in[1];
  const float* edge_same = (const float*)d_in[2];
  const int*   ei        = (const int*)d_in[4];
  const float* adapt_w   = (const float*)d_in[5];
  const float* adapt_b   = (const float*)d_in[6];
  const float* ln0_g     = (const float*)d_in[7];
  const float* ln0_b     = (const float*)d_in[8];
  const float* lng       = (const float*)d_in[9];
  const float* lnb       = (const float*)d_in[10];
  const float* Wt        = (const float*)d_in[11];
  const float* bt        = (const float*)d_in[12];
  const float* Wks       = (const float*)d_in[13];
  const float* bks       = (const float*)d_in[14];
  const float* Wkd       = (const float*)d_in[15];
  const float* bkd       = (const float*)d_in[16];
  const float* Wq        = (const float*)d_in[17];
  const float* bq        = (const float*)d_in[18];
  const float* Wvs       = (const float*)d_in[19];
  const float* bvs       = (const float*)d_in[20];
  const float* Wvd       = (const float*)d_in[21];
  const float* bvd       = (const float*)d_in[22];

  const int nN = in_sizes[3];   // x_time has N elements
  const int nE = in_sizes[1];   // edge_time has E elements

  char* ws = (char*)d_ws;
  size_t off = 0;
  auto carve = [&](size_t bytes) -> void* {
    void* p = ws + off;
    off += (bytes + 255) & ~(size_t)255;
    return p;
  };
  _Float16* hn16  = (_Float16*)carve((size_t)nN * 64 * sizeof(_Float16));
  float*    qb    = (float*)carve((size_t)nN * 64 * sizeof(float));
  float*    attb  = (float*)carve((size_t)nE * sizeof(float));
  float*    mb    = (float*)carve((size_t)nN * sizeof(float));
  float*    denb  = (float*)carve((size_t)nN * sizeof(float));
  float*    aggrb = (float*)carve((size_t)nN * 64 * sizeof(float));
  _Float16* aT    = (_Float16*)carve(64 * 128 * sizeof(_Float16));
  _Float16 *wtT[2], *wksT[2], *wkdT[2], *wvsT[2], *wvdT[2], *wqT[2];
  for (int l = 0; l < 2; ++l) {
    wtT[l]  = (_Float16*)carve(64 * 72 * sizeof(_Float16));
    wksT[l] = (_Float16*)carve(64 * 64 * sizeof(_Float16));
    wkdT[l] = (_Float16*)carve(64 * 64 * sizeof(_Float16));
    wvsT[l] = (_Float16*)carve(64 * 64 * sizeof(_Float16));
    wvdT[l] = (_Float16*)carve(64 * 64 * sizeof(_Float16));
    wqT[l]  = (_Float16*)carve(64 * 64 * sizeof(_Float16));
  }

  float* h = (float*)d_out;  // h lives in the output buffer (final layer result)

  auto Tpose = [&](const float* W, _Float16* out, int K, int stride) {
    int total = 64 * stride;
    transpose_w_kernel<<<(total + 255) / 256, 256, 0, stream>>>(W, out, K, 64, stride);
  };
  Tpose(adapt_w, aT, 128, 128);
  for (int l = 0; l < 2; ++l) {
    Tpose(Wt + (size_t)l * 65 * 64, wtT[l], 65, 72);
    Tpose(Wks + (size_t)l * 4096, wksT[l], 64, 64);
    Tpose(Wkd + (size_t)l * 4096, wkdT[l], 64, 64);
    Tpose(Wvs + (size_t)l * 4096, wvsT[l], 64, 64);
    Tpose(Wvd + (size_t)l * 4096, wvdT[l], 64, 64);
    Tpose(Wq  + (size_t)l * 4096, wqT[l],  64, 64);
  }

  const int ntN = (nN + 15) / 16;
  const int nbN = (ntN + WPB - 1) / WPB;
  n0_kernel<<<nbN, 256, 0, stream>>>(x, aT, adapt_b, ln0_g, ln0_b, h, nN, ntN);

  const int ntE = (nE + 15) / 16;
  const int ebk = 1024;  // persistent edge blocks
  for (int l = 0; l < 2; ++l) {
    n1_kernel<<<nbN, 256, 0, stream>>>(h, lng + l * 64, lnb + l * 64, wqT[l],
                                       bq + l * 64, hn16, qb, nN, ntN);
    init_kernel<<<(nN * 64 + 255) / 256, 256, 0, stream>>>(mb, denb, aggrb, nN);
    edge_kernel<1><<<ebk, 256, 0, stream>>>(
        edge_time, edge_same, ei, ei + nE, hn16, qb,
        wtT[l], wksT[l], wkdT[l], bt + l * 64, bks + l * 64, bkd + l * 64,
        attb, mb, denb, aggrb, nE, ntE);
    edge_kernel<2><<<ebk, 256, 0, stream>>>(
        edge_time, edge_same, ei, ei + nE, hn16, qb,
        wtT[l], wvsT[l], wvdT[l], bt + l * 64, bvs + l * 64, bvd + l * 64,
        attb, mb, denb, aggrb, nE, ntE);
    n2_kernel<<<(nN * 64 + 255) / 256, 256, 0, stream>>>(h, aggrb, denb, nN);
  }
}
